// GNN_72181220376682
// MI455X (gfx1250) — matrix operand order repaired and checked by reading, not compiled, not run on hardware
//
#include <hip/hip_runtime.h>
#include <hip/hip_bf16.h>
#include <stdint.h>
#include <stddef.h>

// ---------------------------------------------------------------------------
// CDNA5 / gfx1250 GCN pipeline.
//   GEMMs: v_wmma_f32_16x16x32_bf16 (wave32, 16x16 C tile, K-step 32,
//          4 accumulators per wave -> 16x64 strip, A-frag reuse x4)
//   Edge propagate: b128 L2-resident gather + 4x global_atomic_add_f32 scatter
//   BN stats: global_atomic_add_f64
// ---------------------------------------------------------------------------

typedef __attribute__((ext_vector_type(16))) __bf16       v16bf;
typedef __attribute__((ext_vector_type(8)))  float        v8f;
typedef __attribute__((ext_vector_type(4)))  unsigned int uv4;
typedef __attribute__((ext_vector_type(2)))  unsigned int uv2;
typedef __attribute__((ext_vector_type(4)))  float        fv4;

union FragBF { uv4 q[2]; v16bf v; };

__device__ __forceinline__ unsigned short f2b_rne(float f) {
  unsigned int u = __float_as_uint(f);
  unsigned int r = u + 0x7FFFu + ((u >> 16) & 1u);
  return (unsigned short)(r >> 16);
}

// ---------------- elementwise helpers ----------------

// n must be a multiple of 4 (always true here: multiples of 128).
__global__ void f32_to_bf16_k(const float* __restrict__ s,
                              unsigned short* __restrict__ d, size_t n4) {
  size_t i = (size_t)blockIdx.x * blockDim.x + threadIdx.x;
  size_t st = (size_t)gridDim.x * blockDim.x;
  for (; i < n4; i += st) {
    fv4 v = *(const fv4*)(s + i * 4);
    uv2 o;
    o.x = (unsigned)f2b_rne(v.x) | ((unsigned)f2b_rne(v.y) << 16);
    o.y = (unsigned)f2b_rne(v.z) | ((unsigned)f2b_rne(v.w) << 16);
    *(uv2*)(d + i * 4) = o;
  }
}

__global__ void fill_f32_k(float* __restrict__ p, float v, size_t n) {
  size_t i = (size_t)blockIdx.x * blockDim.x + threadIdx.x;
  size_t st = (size_t)gridDim.x * blockDim.x;
  for (; i < n; i += st) p[i] = v;
}

__global__ void zero_f64_k(double* __restrict__ p, int n) {
  int i = blockIdx.x * blockDim.x + threadIdx.x;
  if (i < n) p[i] = 0.0;
}

// dst[n][D] = b[D], vectorized by 4 (D multiple of 4).
__global__ void fill_bias_k(float* __restrict__ dst, const float* __restrict__ b,
                            size_t n, int D) {
  int Q = D >> 2;
  size_t i = (size_t)blockIdx.x * blockDim.x + threadIdx.x;
  size_t st = (size_t)gridDim.x * blockDim.x;
  size_t tot = n * (size_t)Q;
  for (; i < tot; i += st) {
    fv4 v = *(const fv4*)(b + ((int)(i % (size_t)Q) << 2));
    *(fv4*)(dst + i * 4) = v;
  }
}

// ---------------- weight packing into WMMA B-fragment layout ----------------
// Layout (contiguous K order, same as the A loads; bf16 B 32x16): tile jt (16 cols), k-step ks (32 K),
// lane l holds column n = jt*16+(l&15), elements e=0..15 -> K = ks*32 + (l>>4)*16 + e.
// Stored contiguously: P[((jt*KS + ks)*32 + l)*16 + e]
__global__ void pack_w_k(const float* __restrict__ W, unsigned short* __restrict__ P,
                         int K, int Nout) {
  int KS = K >> 5;
  int total = K * Nout;
  for (int o = blockIdx.x * blockDim.x + threadIdx.x; o < total;
       o += gridDim.x * blockDim.x) {
    int e  = o & 15;
    int l  = (o >> 4) & 31;
    int ks = (o >> 9) % KS;
    int jt = (o >> 9) / KS;
    int nn = jt * 16 + (l & 15);
    int kk = ks * 32 + (l >> 4) * 16 + e;
    P[o] = f2b_rne(W[(size_t)kk * Nout + nn]);
  }
}

// ---------------- WMMA GEMM: C[n,Nout] = A[n,K](bf16) * Wpacked + bias ------
// Block = 256 threads = 8 waves; each wave computes a 16-row x (16*NT)-col strip.
template<int NT, int ADD_BIAS, int RELU>
__global__ void __launch_bounds__(256)
gemm_wmma_k(const unsigned short* __restrict__ A,
            const unsigned short* __restrict__ Bp,
            const float* __restrict__ bias,
            float* __restrict__ Cout,
            int n, int K, int Nout) {
  const int lane = threadIdx.x & 31;
  const int wave = threadIdx.x >> 5;
  const int rowTiles = (n + 15) >> 4;
  const int rowTile  = blockIdx.x * 8 + wave;
  if (rowTile >= rowTiles) return;               // wave-uniform exit (EXEC all-1 in WMMA)
  const int rowBase = rowTile << 4;
  const int KS = K >> 5;

  // A fragment addressing: lane -> row m = rowBase+(lane&15), kb = (lane>>4)*16
  const int m  = rowBase + (lane & 15);
  const int kb = (lane >> 4) << 4;
  const unsigned short* arow = A + (size_t)m * K + kb;

  v8f acc[NT];
#pragma unroll
  for (int j = 0; j < NT; ++j) acc[j] = (v8f){};

  for (int ks = 0; ks < KS; ++ks) {
    FragBF af;
    af.q[0] = *(const uv4*)(arow + ks * 32);        // K = kb .. kb+7
    af.q[1] = *(const uv4*)(arow + ks * 32 + 8);    // K = kb+8 .. kb+15
#pragma unroll
    for (int j = 0; j < NT; ++j) {
      const int jt = blockIdx.y * NT + j;
      const unsigned short* bp =
          Bp + ((size_t)(jt * KS + ks) * 32 + lane) * 16;
      FragBF bf;
      bf.q[0] = *(const uv4*)(bp);
      bf.q[1] = *(const uv4*)(bp + 8);
      acc[j] = __builtin_amdgcn_wmma_f32_16x16x32_bf16(
          false, af.v, false, bf.v, (short)0, acc[j], false, false);
    }
  }

  // C layout: lane l, vgpr r -> (M = rowBase + r + 8*(l>>4), N = colBase + (l&15))
  const int mrow0 = rowBase + ((lane >> 4) << 3);
#pragma unroll
  for (int j = 0; j < NT; ++j) {
    const int nCol = (blockIdx.y * NT + j) * 16 + (lane & 15);
    float bv = ADD_BIAS ? bias[nCol] : 0.0f;
#pragma unroll
    for (int r = 0; r < 8; ++r) {
      float v = acc[j][r] + bv;
      if (RELU) v = fmaxf(v, 0.0f);
      Cout[(size_t)(mrow0 + r) * Nout + nCol] = v;
    }
  }
}

// ---------------- GCN normalization precompute ----------------

__global__ void deg_edges_k(const int* __restrict__ col, const float* __restrict__ ew,
                            float* __restrict__ deg, int E_) {
  int i = blockIdx.x * blockDim.x + threadIdx.x;
  if (i < E_) atomicAdd(&deg[col[i]], ew[i]);
}

__global__ void dis_k(float* __restrict__ deg, int n) {
  int i = blockIdx.x * blockDim.x + threadIdx.x;
  if (i < n) {
    float d = deg[i];
    deg[i] = (d > 0.0f) ? rsqrtf(fmaxf(d, 1e-12f)) : 0.0f;
  }
}

__global__ void norm_k(const int* __restrict__ row, const int* __restrict__ col,
                       const float* __restrict__ ew, const float* __restrict__ dis,
                       float* __restrict__ nrm, int E_, int N_) {
  int i = blockIdx.x * blockDim.x + threadIdx.x;
  if (i < E_ + N_) {
    if (i < E_) nrm[i] = dis[row[i]] * ew[i] * dis[col[i]];
    else { float d = dis[i - E_]; nrm[i] = d * d; }
  }
}

// ---------------- edge propagate: dst[col] += src[row] * norm ----------------
// Vectorized by 4: one b128 gather (L2-resident, src <= 102MB < 192MB L2) and
// four consecutive global_atomic_add_f32 per thread. Consecutive threads cover
// consecutive feature quads of one edge -> fully coalesced.
__global__ void propagate_k(const float* __restrict__ src, float* __restrict__ dst,
                            const int* __restrict__ row, const int* __restrict__ col,
                            const float* __restrict__ nrm, int E_, int N_, int D) {
  const int Q = D >> 2;                       // quads per edge row
  size_t tot = (size_t)(E_ + N_) * (size_t)Q;
  size_t st  = (size_t)gridDim.x * blockDim.x;
  for (size_t t = (size_t)blockIdx.x * blockDim.x + threadIdx.x; t < tot; t += st) {
    int e = (int)(t / (size_t)Q);
    int c = ((int)(t % (size_t)Q)) << 2;
    int r, cn;
    if (e < E_) { r = row[e]; cn = col[e]; }
    else        { r = cn = e - E_; }
    float w = nrm[e];
    fv4 v = *(const fv4*)(src + (size_t)r * D + c);
    float* dp = dst + (size_t)cn * D + c;
    atomicAdd(dp + 0, v.x * w);
    atomicAdd(dp + 1, v.y * w);
    atomicAdd(dp + 2, v.z * w);
    atomicAdd(dp + 3, v.w * w);
  }
}

// ---------------- BatchNorm (training-mode batch stats) ----------------

__global__ void bn_stats_k(const float* __restrict__ a, double* __restrict__ st,
                           int n, int D) {
  int c = threadIdx.x;                              // blockDim.x == D
  double s = 0.0, ss = 0.0;
  for (int r = blockIdx.x; r < n; r += gridDim.x) {
    float v = a[(size_t)r * D + c];
    s += (double)v; ss += (double)v * (double)v;
  }
  atomicAdd(&st[c], s);                             // global_atomic_add_f64
  atomicAdd(&st[D + c], ss);
}

template<int RELU>
__global__ void bn_apply_k(const float* __restrict__ a, const double* __restrict__ st,
                           const float* __restrict__ gamma, const float* __restrict__ beta,
                           unsigned short* __restrict__ obf, int n, int D) {
  int c = threadIdx.x;                              // blockDim.x == D
  float mean  = (float)(st[c] / (double)n);
  float var   = (float)(st[D + c] / (double)n) - mean * mean;
  float scale = gamma[c] * rsqrtf(var + 1e-5f);
  float shift = beta[c] - mean * scale;
  for (int r = blockIdx.x; r < n; r += gridDim.x) {
    float v = a[(size_t)r * D + c] * scale + shift;
    if (RELU) v = fmaxf(v, 0.0f);
    obf[(size_t)r * D + c] = f2b_rne(v);
  }
}

// ---------------------------------------------------------------------------

extern "C" void kernel_launch(void* const* d_in, const int* in_sizes, int n_in,
                              void* d_out, int out_size, void* d_ws, size_t ws_size,
                              hipStream_t stream) {
  const int F = 128, H = 128;
  const int N  = in_sizes[0] / F;       // 100000
  const int E  = in_sizes[2];           // 1600000
  const int Dmax = 2 * H;               // 256

  const float* x  = (const float*)d_in[0];
  const int*   ei = (const int*)d_in[1];
  const float* ew = (const float*)d_in[2];
  const int* row = ei;
  const int* col = ei + E;

  // param layers: {W,b,gamma,beta} x {pre0,pre1,conv0,conv1,conv2,post0,post1} + clf
  struct Layer { const float *W, *b, *g, *be; int K, Nout; };
  Layer L[8];
  int dims[8][2] = {{128,128},{128,128},{128,256},{256,256},{256,128},
                    {128,128},{128,128},{128,16}};
  for (int i = 0; i < 7; ++i) {
    L[i].W  = (const float*)d_in[3 + 4*i + 0];
    L[i].b  = (const float*)d_in[3 + 4*i + 1];
    L[i].g  = (const float*)d_in[3 + 4*i + 2];
    L[i].be = (const float*)d_in[3 + 4*i + 3];
    L[i].K = dims[i][0]; L[i].Nout = dims[i][1];
  }
  L[7].W = (const float*)d_in[31]; L[7].b = (const float*)d_in[32];
  L[7].g = nullptr; L[7].be = nullptr; L[7].K = 128; L[7].Nout = 16;

  // ---- workspace carve-out ----
  auto alignup = [](size_t v) { return (v + 255) & ~(size_t)255; };
  char* ws = (char*)d_ws;
  size_t off = 0;
  unsigned short* actA = (unsigned short*)(ws + off); off = alignup(off + (size_t)N * Dmax * 2);
  float*  tmpF  = (float*)(ws + off);  off = alignup(off + (size_t)N * Dmax * 4);
  float*  tmpG  = (float*)(ws + off);  off = alignup(off + (size_t)N * Dmax * 4);
  float*  dis   = (float*)(ws + off);  off = alignup(off + (size_t)N * 4);
  float*  normv = (float*)(ws + off);  off = alignup(off + (size_t)(E + N) * 4);
  double* dstats = (double*)(ws + off); off = alignup(off + 2 * (size_t)Dmax * 8);
  unsigned short* packW[8];
  for (int i = 0; i < 8; ++i) {
    packW[i] = (unsigned short*)(ws + off);
    off = alignup(off + (size_t)L[i].K * L[i].Nout * 2);
  }
  (void)ws_size;

  auto blocks = [](size_t n, int b) { return (unsigned)((n + (size_t)b - 1) / b); };
  const int rowTiles = (N + 15) >> 4;
  const unsigned gxGemm = (unsigned)((rowTiles + 7) / 8);

  // ---- 0. pack weights (bf16 WMMA fragment layout), convert x -> bf16 ----
  for (int i = 0; i < 8; ++i) {
    size_t tot = (size_t)L[i].K * L[i].Nout;
    pack_w_k<<<blocks(tot, 256), 256, 0, stream>>>(L[i].W, packW[i], L[i].K, L[i].Nout);
  }
  f32_to_bf16_k<<<blocks((size_t)N * F / 4, 256), 256, 0, stream>>>(x, actA,
                                                                    (size_t)N * F / 4);

  // ---- 1. GCN normalization (shared by all three convs) ----
  fill_f32_k<<<blocks((size_t)N, 256), 256, 0, stream>>>(dis, 1.0f, (size_t)N); // self-loop deg
  deg_edges_k<<<blocks((size_t)E, 256), 256, 0, stream>>>(col, ew, dis, E);
  dis_k<<<blocks((size_t)N, 256), 256, 0, stream>>>(dis, N);
  norm_k<<<blocks((size_t)(E + N), 256), 256, 0, stream>>>(row, col, ew, dis, normv, E, N);

  // ---- MLP block: actA -> relu(actA@W + b) -> BN -> actA (bf16) ----
  auto mlp_block = [&](const Layer& p, const unsigned short* pw) {
    dim3 g(gxGemm, (unsigned)(p.Nout / 64));
    gemm_wmma_k<4, 1, 1><<<g, 256, 0, stream>>>(actA, pw, p.b, tmpF, N, p.K, p.Nout);
    zero_f64_k<<<2, 256, 0, stream>>>(dstats, 2 * p.Nout);
    bn_stats_k<<<1024, p.Nout, 0, stream>>>(tmpF, dstats, N, p.Nout);
    bn_apply_k<0><<<2048, p.Nout, 0, stream>>>(tmpF, dstats, p.g, p.be, actA, N, p.Nout);
  };

  // ---- execute pipeline ----

  // pre blocks
  for (int i = 0; i < 2; ++i) mlp_block(L[i], packW[i]);

  // conv blocks: gemm -> (bias-init + scatter) -> BN -> ReLU -> actA
  for (int i = 2; i < 5; ++i) {
    const Layer& p = L[i];
    dim3 g(gxGemm, (unsigned)(p.Nout / 64));
    gemm_wmma_k<4, 0, 0><<<g, 256, 0, stream>>>(actA, packW[i], nullptr, tmpF, N, p.K, p.Nout);
    fill_bias_k<<<blocks((size_t)N * p.Nout / 4, 256), 256, 0, stream>>>(
        tmpG, p.b, (size_t)N, p.Nout);
    size_t tot = (size_t)(E + N) * (size_t)(p.Nout / 4);
    propagate_k<<<blocks(tot, 256), 256, 0, stream>>>(tmpF, tmpG, row, col, normv,
                                                      E, N, p.Nout);
    zero_f64_k<<<2, 256, 0, stream>>>(dstats, 2 * p.Nout);
    bn_stats_k<<<1024, p.Nout, 0, stream>>>(tmpG, dstats, N, p.Nout);
    bn_apply_k<1><<<2048, p.Nout, 0, stream>>>(tmpG, dstats, p.g, p.be, actA, N, p.Nout);
  }

  // post blocks
  for (int i = 5; i < 7; ++i) mlp_block(L[i], packW[i]);

  // classifier: d_out[N,16] = actA @ clf_W + clf_b  (fp32 out)
  {
    dim3 g(gxGemm, 1);
    gemm_wmma_k<1, 1, 0><<<g, 256, 0, stream>>>(actA, packW[7], L[7].b,
                                                (float*)d_out, N, 128, 16);
  }
  (void)out_size; (void)n_in;
}
